// align_unif_61375082660447
// MI455X (gfx1250) — matrix operand order repaired and checked by reading, not compiled, not run on hardware
//
#include <hip/hip_runtime.h>
#include <hip/hip_bf16.h>
#include <math.h>

// align+uniformity loss for x,y in [8192, 256] f32.
//   align = mean_i ||x_i - y_i||^2                      (ALPHA = 2)
//   unif  = log( (S - n) / (n*(n-1)) ),  S = sum_ij exp(-2*||x_i-x_j||^2)
// Gram matrix x @ x^T via V_WMMA_F32_16X16X32_BF16 with f32 -> bf16 hi+lo
// split: dot ~= hi*hi + hi*lo + lo*hi (3 WMMAs per K=32 chunk).
// Per-wave tile 32Mx64N (8 acc tiles); block tile 128x128 (8 waves, 4Mx2N).
// All loop trip counts are compile-time so every global load uses a
// precomputed base pointer + immediate offset (no per-iteration address VALU).

typedef __attribute__((ext_vector_type(16))) __bf16          v16bf;
typedef __attribute__((ext_vector_type(8)))  float           v8f;
typedef __attribute__((ext_vector_type(8)))  unsigned short  u16x8;

#define N_ROWS 8192
#define DIM    256
#define BT     128   // block tile (both M and N)
#define WAVES  8     // 256 threads, wave32

union FragBF {
    v16bf v;
    u16x8 h[2];   // h[0] = bf16 slots 0..7, h[1] = slots 8..15
};

// ---------------------------------------------------------------------------
// Pass 1: per-row prep. bf16 hi/lo split of x, row norms sq[i] = ||x_i||^2,
// per-row align contribution ||x_i - y_i||^2. One 256-thread block per row.
// ---------------------------------------------------------------------------
__global__ __launch_bounds__(256) void prep_kernel(
    const float* __restrict__ x, const float* __restrict__ y,
    unsigned short* __restrict__ xh, unsigned short* __restrict__ xl,
    float* __restrict__ sq, float* __restrict__ alignRows)
{
    const int row = blockIdx.x;
    const int tid = threadIdx.x;
    const size_t idx = (size_t)row * DIM + tid;

    const float xv = x[idx];
    const float yv = y[idx];

    // truncating split: hi keeps top 16 bits, lo = residual truncated to bf16
    const unsigned int bits = __float_as_uint(xv);
    const float hif = __uint_as_float(bits & 0xFFFF0000u);
    const float lof = xv - hif;
    xh[idx] = (unsigned short)(bits >> 16);
    xl[idx] = (unsigned short)(__float_as_uint(lof) >> 16);

    float s1 = xv * xv;                 // -> sq
    const float d = xv - yv;
    float s2 = d * d;                   // -> align

    for (int o = 16; o > 0; o >>= 1) {
        s1 += __shfl_xor(s1, o, 32);
        s2 += __shfl_xor(s2, o, 32);
    }
    __shared__ float l1[WAVES], l2[WAVES];
    const int lane = tid & 31, wave = tid >> 5;
    if (lane == 0) { l1[wave] = s1; l2[wave] = s2; }
    __syncthreads();
    if (tid == 0) {
        float a = 0.f, b = 0.f;
        for (int w = 0; w < WAVES; ++w) { a += l1[w]; b += l2[w]; }
        sq[row] = a;
        alignRows[row] = b;
    }
}

// ---------------------------------------------------------------------------
// Pass 2: tiled Gram matrix + exp epilogue.
// ---------------------------------------------------------------------------
__global__ __launch_bounds__(256) void gram_unif_kernel(
    const unsigned short* __restrict__ xh,
    const unsigned short* __restrict__ xl,
    const float* __restrict__ sq,
    float* __restrict__ sPart)
{
    const int lane = threadIdx.x & 31;
    const int wave = threadIdx.x >> 5;
    const int half = lane >> 4;       // lane group 0-15 vs 16-31
    const int l16  = lane & 15;

    const int mBase = blockIdx.y * BT + (wave & 3) * 32;  // 2 strips of 16
    const int nBase = blockIdx.x * BT + (wave >> 2) * 64; // 4 tiles of 16

    // 8 accumulator tiles: acc[s*4 + t], strip s in {0,1}, n-tile t in {0..3}
    v8f acc[8];
#pragma unroll
    for (int i = 0; i < 8; ++i)
#pragma unroll
        for (int j = 0; j < 8; ++j) acc[i][j] = 0.0f;

    // A fragment base pointers (16x32 bf16 ISA layout): row = strip base + l16,
    //   slots 0..7  = K[kb + half*8 .. +7], slots 8..15 = K[kb+16+half*8 .. +7]
    const unsigned short* pAh[2];
    const unsigned short* pAl[2];
#pragma unroll
    for (int s = 0; s < 2; ++s) {
        const size_t off = (size_t)(mBase + s * 16 + l16) * DIM + half * 8;
        pAh[s] = xh + off;
        pAl[s] = xl + off;
    }
    // B fragment base pointers (32x16 bf16): col = nBase + t*16 + l16,
    //   slots 0..7 = K[kb + half*8 .. +7], slots 8..15 = K[kb+16+half*8 .. +7] of row(col) of x  (B = x^T)
    const unsigned short* pBh[4];
    const unsigned short* pBl[4];
#pragma unroll
    for (int t = 0; t < 4; ++t) {
        const size_t off = (size_t)(nBase + t * 16 + l16) * DIM + half * 8;
        pBh[t] = xh + off;
        pBl[t] = xl + off;
    }

#pragma unroll
    for (int kc = 0; kc < DIM / 32; ++kc) {
        const int kb = kc * 32;   // compile-time => immediate load offsets

        FragBF ah[2], al[2];
#pragma unroll
        for (int s = 0; s < 2; ++s) {
            ah[s].h[0] = *(const u16x8*)(pAh[s] + kb);
            ah[s].h[1] = *(const u16x8*)(pAh[s] + kb + 16);
            al[s].h[0] = *(const u16x8*)(pAl[s] + kb);
            al[s].h[1] = *(const u16x8*)(pAl[s] + kb + 16);
        }

#pragma unroll
        for (int t = 0; t < 4; ++t) {
            FragBF bh, bl;
            bh.h[0] = *(const u16x8*)(pBh[t] + kb);
            bh.h[1] = *(const u16x8*)(pBh[t] + kb + 16);
            bl.h[0] = *(const u16x8*)(pBl[t] + kb);
            bl.h[1] = *(const u16x8*)(pBl[t] + kb + 16);
#pragma unroll
            for (int s = 0; s < 2; ++s) {
                const int a = s * 4 + t;
                // hi*hi + hi*lo + lo*hi  (lo*lo ~2^-16 rel, dropped)
                acc[a] = __builtin_amdgcn_wmma_f32_16x16x32_bf16(
                    false, ah[s].v, false, bh.v, (short)0, acc[a], false, false);
                acc[a] = __builtin_amdgcn_wmma_f32_16x16x32_bf16(
                    false, ah[s].v, false, bl.v, (short)0, acc[a], false, false);
                acc[a] = __builtin_amdgcn_wmma_f32_16x16x32_bf16(
                    false, al[s].v, false, bh.v, (short)0, acc[a], false, false);
            }
        }
    }

    // Epilogue. C/D layout: VGPR v -> M = stripBase + v + 8*half,
    //                       lane  -> N = nBase + t*16 + l16
    float sqm[2][8];
#pragma unroll
    for (int s = 0; s < 2; ++s)
#pragma unroll
        for (int v = 0; v < 8; ++v)
            sqm[s][v] = sq[mBase + s * 16 + v + 8 * half];

    float ssum = 0.0f;
#pragma unroll
    for (int t = 0; t < 4; ++t) {
        const float sqn = sq[nBase + t * 16 + l16];
#pragma unroll
        for (int s = 0; s < 2; ++s) {
#pragma unroll
            for (int v = 0; v < 8; ++v) {
                float d2 = sqm[s][v] + sqn - 2.0f * acc[s * 4 + t][v];
                d2 = fmaxf(d2, 0.0f);
                ssum += __expf(-2.0f * d2);   // T = 2
            }
        }
    }

    for (int o = 16; o > 0; o >>= 1) ssum += __shfl_xor(ssum, o, 32);
    __shared__ float ls[WAVES];
    if (lane == 0) ls[wave] = ssum;
    __syncthreads();
    if (threadIdx.x == 0) {
        float tot = 0.f;
        for (int w = 0; w < WAVES; ++w) tot += ls[w];
        sPart[(size_t)blockIdx.y * gridDim.x + blockIdx.x] = tot;
    }
}

// ---------------------------------------------------------------------------
// Pass 3: deterministic final reduction (single block, fp64 accumulation).
// ---------------------------------------------------------------------------
__global__ __launch_bounds__(256) void finalize_kernel(
    const float* __restrict__ alignRows, const float* __restrict__ sPart,
    int nRows, int nPart, float* __restrict__ out)
{
    double a = 0.0, s = 0.0;
    for (int i = threadIdx.x; i < nRows; i += 256) a += (double)alignRows[i];
    for (int i = threadIdx.x; i < nPart; i += 256) s += (double)sPart[i];

    __shared__ double la[256], lsum[256];
    la[threadIdx.x] = a; lsum[threadIdx.x] = s;
    __syncthreads();
    for (int st = 128; st > 0; st >>= 1) {
        if (threadIdx.x < st) {
            la[threadIdx.x]   += la[threadIdx.x + st];
            lsum[threadIdx.x] += lsum[threadIdx.x + st];
        }
        __syncthreads();
    }
    if (threadIdx.x == 0) {
        const double n = (double)nRows;
        const double align = la[0] / n;
        const double S = lsum[0];
        const double meanVal = (S - n) / (n * (n - 1.0));
        const double unif = log(meanVal);
        out[0] = (float)(align + unif);
        out[1] = (float)align;
        out[2] = (float)unif;
    }
}

extern "C" void kernel_launch(void* const* d_in, const int* in_sizes, int n_in,
                              void* d_out, int out_size, void* d_ws, size_t ws_size,
                              hipStream_t stream) {
    const float* x = (const float*)d_in[0];
    const float* y = (const float*)d_in[1];
    float* out = (float*)d_out;

    // workspace layout (needs ~8.3 MB):
    //   xh: 8192*256 bf16 (4 MB) | xl: 4 MB | sq: 32 KB | alignRows: 32 KB
    //   sPart: (8192/128)^2 = 4096 floats (16 KB)
    char* ws = (char*)d_ws;
    unsigned short* xh = (unsigned short*)(ws);
    unsigned short* xl = (unsigned short*)(ws + (size_t)N_ROWS * DIM * 2);
    float* sq          = (float*)(ws + (size_t)N_ROWS * DIM * 4);
    float* alignRows   = sq + N_ROWS;
    float* sPart       = alignRows + N_ROWS;

    const int g = N_ROWS / BT;   // 64

    prep_kernel<<<N_ROWS, 256, 0, stream>>>(x, y, xh, xl, sq, alignRows);
    gram_unif_kernel<<<dim3(g, g), 256, 0, stream>>>(xh, xl, sq, sPart);
    finalize_kernel<<<1, 256, 0, stream>>>(alignRows, sPart, N_ROWS, g * g, out);
}
